// BinaryDecisionNetwork_67224828117016
// MI455X (gfx1250) — matrix operand order repaired and checked by reading, not compiled, not run on hardware
//
#include <hip/hip_runtime.h>
#include <hip/hip_bf16.h>
#include <math.h>

typedef __attribute__((ext_vector_type(16))) _Float16 v16h;
typedef __attribute__((ext_vector_type(8)))  _Float16 v8h;
typedef __attribute__((ext_vector_type(8)))  float    v8f;

#define LN_  128
#define CS_  32

// A operand (16x32 f16, M rows in lanes, K striped): halfs 0..7 at p, 8..15 at p+16
__device__ __forceinline__ v16h make_a(const _Float16* p) {
  v8h lo = *(const v8h*)p;
  v8h hi = *(const v8h*)(p + 16);
  v16h r;
#pragma unroll
  for (int i = 0; i < 8; ++i) { r[i] = lo[i]; r[i + 8] = hi[i]; }
  return r;
}
// B operand (32x16 f16, N cols in lanes): halfs 0..7 at p, 8..15 at p+16
__device__ __forceinline__ v16h make_b(const _Float16* p) {
  v8h lo = *(const v8h*)p;
  v8h hi = *(const v8h*)(p + 16);
  v16h r;
#pragma unroll
  for (int i = 0; i < 8; ++i) { r[i] = lo[i]; r[i + 8] = hi[i]; }
  return r;
}

// CDNA5 async DMA: global -> LDS, 16 bytes per lane, tracked by ASYNCcnt.
// VGPR[vdst] = wave-relative LDS byte offset (low 32 bits of the flat shared
// address), VGPR pair = 64-bit global address (GV mode, saddr = off).
__device__ __forceinline__ void async_copy_b128(const float* gsrc, float* lds_dst) {
  unsigned ldsa = (unsigned)(size_t)lds_dst;          // addr[31:0] == LDS offset
  unsigned long long ga = (unsigned long long)(size_t)gsrc;
  asm volatile("global_load_async_to_lds_b128 %0, %1, off"
               :: "v"(ldsa), "v"(ga) : "memory");
}
__device__ __forceinline__ void wait_async0() {
  asm volatile("s_wait_asynccnt 0x0" ::: "memory");
}

__global__ __launch_bounds__(32) void fused_bdn_kernel(
    const float* __restrict__ key_, const float* __restrict__ semantic,
    const float* __restrict__ knowledge, const float* __restrict__ mapping,
    const float* __restrict__ orig,
    const float* __restrict__ kp_dw_w, const float* __restrict__ kp_dw_b,
    const float* __restrict__ kp_pw_w, const float* __restrict__ kp_pw_b,
    const float* __restrict__ q_w, const float* __restrict__ q_b,
    const float* __restrict__ k_w, const float* __restrict__ k_b,
    const float* __restrict__ v_w, const float* __restrict__ v_b,
    const float* __restrict__ o_w, const float* __restrict__ o_b,
    const float* __restrict__ ln_w, const float* __restrict__ ln_b,
    const float* __restrict__ sem_w, const float* __restrict__ sem_b,
    const float* __restrict__ kr_w, const float* __restrict__ kr_b,
    const float* __restrict__ mr_w, const float* __restrict__ mr_b,
    const float* __restrict__ kc_dw_w, const float* __restrict__ kc_dw_b,
    const float* __restrict__ kc_pw_w, const float* __restrict__ kc_pw_b,
    const float* __restrict__ op_w, const float* __restrict__ op_b,
    const float* __restrict__ ap_w, const float* __restrict__ ap_b,
    const float* __restrict__ ol1_w, const float* __restrict__ ol1_b,
    const float* __restrict__ ol2_w, const float* __restrict__ ol2_b,
    float* __restrict__ out)
{
  // LDS staging (~93 KB / workgroup; WGP budget 320 KB)
  __shared__ __align__(16) _Float16 sXt[LN_ * CS_];        // semantic, [l][c] f16
  __shared__ __align__(16) _Float16 sQt[LN_ * CS_ + 64];   // Q, [l][c] f16 (+zero pad)
  __shared__ __align__(16) _Float16 sKP[CS_ * LN_];        // elu(K)+1, [c][l] f16
  __shared__ __align__(16) _Float16 sV [CS_ * LN_];        // V, [c][l] f16
  __shared__ __align__(16) _Float16 sSt[LN_ * CS_];        // att result, [l][c] f16
  __shared__ __align__(16) _Float16 sW [CS_ * CS_];        // current weight, row-major f16
  __shared__ float sCtx[CS_ * 8];                          // ctx[h*8+d][e]
  __shared__ float sS2 [CS_ * LN_];                        // o-proj / LN result f32
  __shared__ float sFeat[CS_ * LN_];                       // fused feats [kp|sem|kc|op]
  __shared__ float sTmp[8 * LN_];                          // kc intermediate
  // async-DMA staged inputs for the late branches (arrive during WMMA phase)
  __shared__ __align__(16) float sKeyF[10 * LN_];          // key    [10][128]
  __shared__ __align__(16) float sKnF [8 * LN_];           // knowledge [8][128]
  __shared__ __align__(16) float sMpF [8 * LN_];           // mapping   [8][128]
  __shared__ __align__(16) float sOrF [LN_];               // original_output [128]

  const int b    = blockIdx.x;
  const int lane = threadIdx.x;
  const int lhi  = lane >> 4;   // 0/1 -> K/M half selector per ISA layouts
  const int llo  = lane & 15;

  // ---- kick off async global->LDS DMA for branch inputs (overlaps WMMAs) ----
  {
    const float* gk = key_      + (size_t)b * 10 * LN_;
    const float* gn = knowledge + (size_t)b * 8 * LN_;
    const float* gm = mapping   + (size_t)b * 8 * LN_;
    const float* go = orig      + (size_t)b * LN_;
#pragma unroll
    for (int i = 0; i < 10; ++i)           // 10*128 floats, 16B per lane-op
      async_copy_b128(gk + (lane + 32 * i) * 4, sKeyF + (lane + 32 * i) * 4);
#pragma unroll
    for (int i = 0; i < 8; ++i) {
      async_copy_b128(gn + (lane + 32 * i) * 4, sKnF + (lane + 32 * i) * 4);
      async_copy_b128(gm + (lane + 32 * i) * 4, sMpF + (lane + 32 * i) * 4);
    }
    async_copy_b128(go + lane * 4, sOrF + lane * 4);
  }

  // ---- stage semantic [32,128] -> LDS transposed f16; zero Qt pad ----
  const float* X = semantic + (size_t)b * CS_ * LN_;
  for (int idx = lane; idx < CS_ * LN_; idx += 32) {
    int c = idx >> 7, l = idx & 127;
    sXt[l * CS_ + c] = (_Float16)X[idx];
  }
  for (int idx = lane; idx < 64; idx += 32) sQt[LN_ * CS_ + idx] = (_Float16)0.f;
  __syncthreads();

  // ================= Q projection: Q = Wq @ X + bq -> sQt ([l][c]) =========
  for (int idx = lane; idx < CS_ * CS_; idx += 32) sW[idx] = (_Float16)q_w[idx];
  __syncthreads();
  for (int mt = 0; mt < 2; ++mt) {
    v16h A = make_a(sW + (mt * 16 + llo) * CS_ + lhi * 8);
    for (int nt = 0; nt < 8; ++nt) {
      v16h Bv = make_b(sXt + (nt * 16 + llo) * CS_ + lhi * 8);
      v8f acc = {};
      acc = __builtin_amdgcn_wmma_f32_16x16x32_f16(false, A, false, Bv, (short)0, acc, false, false);
      int l = nt * 16 + llo;
#pragma unroll
      for (int v = 0; v < 8; ++v) {
        int m = mt * 16 + lhi * 8 + v;
        sQt[l * CS_ + m] = (_Float16)(acc[v] + q_b[m]);
      }
    }
  }
  __syncthreads();

  // ================= K projection -> KP = elu(K)+1 -> sKP ([c][l]) =========
  for (int idx = lane; idx < CS_ * CS_; idx += 32) sW[idx] = (_Float16)k_w[idx];
  __syncthreads();
  for (int mt = 0; mt < 2; ++mt) {
    v16h A = make_a(sW + (mt * 16 + llo) * CS_ + lhi * 8);
    for (int nt = 0; nt < 8; ++nt) {
      v16h Bv = make_b(sXt + (nt * 16 + llo) * CS_ + lhi * 8);
      v8f acc = {};
      acc = __builtin_amdgcn_wmma_f32_16x16x32_f16(false, A, false, Bv, (short)0, acc, false, false);
      int l = nt * 16 + llo;
#pragma unroll
      for (int v = 0; v < 8; ++v) {
        int m = mt * 16 + lhi * 8 + v;
        float x = acc[v] + k_b[m];
        float kp = (x > 0.f) ? (x + 1.f) : __expf(x);   // elu(x)+1
        sKP[m * LN_ + l] = (_Float16)kp;
      }
    }
  }
  __syncthreads();

  // ================= V projection -> sV ([c][l]) ===========================
  for (int idx = lane; idx < CS_ * CS_; idx += 32) sW[idx] = (_Float16)v_w[idx];
  __syncthreads();
  for (int mt = 0; mt < 2; ++mt) {
    v16h A = make_a(sW + (mt * 16 + llo) * CS_ + lhi * 8);
    for (int nt = 0; nt < 8; ++nt) {
      v16h Bv = make_b(sXt + (nt * 16 + llo) * CS_ + lhi * 8);
      v8f acc = {};
      acc = __builtin_amdgcn_wmma_f32_16x16x32_f16(false, A, false, Bv, (short)0, acc, false, false);
      int l = nt * 16 + llo;
#pragma unroll
      for (int v = 0; v < 8; ++v) {
        int m = mt * 16 + lhi * 8 + v;
        sV[m * LN_ + l] = (_Float16)(acc[v] + v_b[m]);
      }
    }
  }
  __syncthreads();

  // ===== context[h] = KP_h @ V_h^T (K=128); two head-pairs, keep diag 8x8 ===
  for (int p = 0; p < 2; ++p) {
    v8f acc = {};
    for (int t = 0; t < 4; ++t) {
      v16h A  = make_a(sKP + (p * 16 + llo) * LN_ + t * 32 + lhi * 8);
      v16h Bv = make_b(sV  + (p * 16 + llo) * LN_ + t * 32 + lhi * 8);
      acc = __builtin_amdgcn_wmma_f32_16x16x32_f16(false, A, false, Bv, (short)0, acc, false, false);
    }
#pragma unroll
    for (int v = 0; v < 8; ++v) {
      int mloc = lhi * 8 + v;
      int nloc = llo;
      if ((mloc >> 3) == (nloc >> 3))                       // same head only
        sCtx[(p * 16 + mloc) * 8 + (nloc & 7)] = acc[v];
    }
  }
  __syncthreads();

  // ===== att: S[e,l] = sum_d ctx[d,e] * Q[d,l]  (K padded to 32) -> sSt =====
  for (int p = 0; p < 2; ++p) {
    v16h A;
#pragma unroll
    for (int j = 0; j < 8; ++j) {
      int k = lhi * 8 + j;                                  // d index 0..15
      float val = ((k >> 3) == (llo >> 3)) ? sCtx[(p * 16 + k) * 8 + (llo & 7)] : 0.f;
      A[j]     = (_Float16)val;
      A[j + 8] = (_Float16)0.f;                             // K=16..31 padding
    }
    for (int nt = 0; nt < 8; ++nt) {
      v16h Bv = make_b(sQt + (nt * 16 + llo) * CS_ + p * 16 + lhi * 8);
      v8f acc = {};
      acc = __builtin_amdgcn_wmma_f32_16x16x32_f16(false, A, false, Bv, (short)0, acc, false, false);
      int l = nt * 16 + llo;
#pragma unroll
      for (int v = 0; v < 8; ++v) {
        int c = p * 16 + lhi * 8 + v;
        sSt[l * CS_ + c] = (_Float16)acc[v];
      }
    }
  }
  __syncthreads();

  // ================= o projection: S2 = Wo @ S + bo -> sS2 f32 ([c][l]) ====
  for (int idx = lane; idx < CS_ * CS_; idx += 32) sW[idx] = (_Float16)o_w[idx];
  __syncthreads();
  for (int mt = 0; mt < 2; ++mt) {
    v16h A = make_a(sW + (mt * 16 + llo) * CS_ + lhi * 8);
    for (int nt = 0; nt < 8; ++nt) {
      v16h Bv = make_b(sSt + (nt * 16 + llo) * CS_ + lhi * 8);
      v8f acc = {};
      acc = __builtin_amdgcn_wmma_f32_16x16x32_f16(false, A, false, Bv, (short)0, acc, false, false);
      int l = nt * 16 + llo;
#pragma unroll
      for (int v = 0; v < 8; ++v) {
        int m = mt * 16 + lhi * 8 + v;
        sS2[m * LN_ + l] = acc[v] + o_b[m];
      }
    }
  }
  __syncthreads();

  // ================= LayerNorm over [32,128] (per batch) ===================
  float s1 = 0.f, s2 = 0.f;
  for (int idx = lane; idx < CS_ * LN_; idx += 32) {
    float x = sS2[idx]; s1 += x; s2 += x * x;
  }
#pragma unroll
  for (int off = 16; off > 0; off >>= 1) {
    s1 += __shfl_xor(s1, off, 32);
    s2 += __shfl_xor(s2, off, 32);
  }
  float mu   = s1 * (1.f / 4096.f);
  float var  = s2 * (1.f / 4096.f) - mu * mu;
  float rstd = rsqrtf(var + 1e-5f);
  for (int idx = lane; idx < CS_ * LN_; idx += 32)
    sS2[idx] = (sS2[idx] - mu) * rstd * ln_w[idx] + ln_b[idx];
  __syncthreads();

  // ================= sem = sem_w[8,32] @ S2n + b -> sFeat rows 8..15 =======
  for (int i = 0; i < 4; ++i) {
    int l = lane + 32 * i;
#pragma unroll
    for (int o = 0; o < 8; ++o) {
      float s = sem_b[o];
      for (int c = 0; c < CS_; ++c) s += sem_w[o * CS_ + c] * sS2[c * LN_ + l];
      sFeat[(8 + o) * LN_ + l] = s;
    }
  }

  // ---- branch inputs were DMA'd during the WMMA phase; drain ASYNCcnt ----
  wait_async0();
  __syncthreads();

  // ================= key branch: dw(K=3) + pw 10->8 -> sFeat rows 0..7 =====
  for (int i = 0; i < 4; ++i) {
    int l = lane + 32 * i;
    float accp[8];
#pragma unroll
    for (int o = 0; o < 8; ++o) accp[o] = kp_pw_b[o];
    for (int c = 0; c < 10; ++c) {
      float x0 = sKeyF[c * LN_ + l];
      float xm = (l > 0)   ? sKeyF[c * LN_ + l - 1] : 0.f;
      float xp = (l < 127) ? sKeyF[c * LN_ + l + 1] : 0.f;
      float d = kp_dw_w[c * 3 + 0] * xm + kp_dw_w[c * 3 + 1] * x0 +
                kp_dw_w[c * 3 + 2] * xp + kp_dw_b[c];
#pragma unroll
      for (int o = 0; o < 8; ++o) accp[o] += kp_pw_w[o * 10 + c] * d;
    }
#pragma unroll
    for (int o = 0; o < 8; ++o) sFeat[o * LN_ + l] = accp[o];
  }

  // ================= knowledge branch -> sFeat rows 16..23 =================
  for (int i = 0; i < 4; ++i) {
    int l = lane + 32 * i;
#pragma unroll
    for (int o = 0; o < 4; ++o) {
      float a = kr_b[o], m2 = mr_b[o];
      for (int c = 0; c < 8; ++c) {
        a  += kr_w[o * 8 + c] * sKnF[c * LN_ + l];
        m2 += mr_w[o * 8 + c] * sMpF[c * LN_ + l];
      }
      sTmp[o * LN_ + l]       = a;
      sTmp[(4 + o) * LN_ + l] = m2;
    }
  }
  __syncthreads();
  for (int i = 0; i < 4; ++i) {
    int l = lane + 32 * i;
    float accp[8];
#pragma unroll
    for (int o = 0; o < 8; ++o) accp[o] = kc_pw_b[o];
#pragma unroll
    for (int c = 0; c < 8; ++c) {
      float x0 = sTmp[c * LN_ + l];
      float xm = (l > 0)   ? sTmp[c * LN_ + l - 1] : 0.f;
      float xp = (l < 127) ? sTmp[c * LN_ + l + 1] : 0.f;
      float d = kc_dw_w[c * 3 + 0] * xm + kc_dw_w[c * 3 + 1] * x0 +
                kc_dw_w[c * 3 + 2] * xp + kc_dw_b[c];
#pragma unroll
      for (int o = 0; o < 8; ++o) accp[o] += kc_pw_w[o * 8 + c] * d;
    }
#pragma unroll
    for (int o = 0; o < 8; ++o) sFeat[(16 + o) * LN_ + l] = fmaxf(accp[o], 0.f);
  }

  // ================= output branch: conv 1->8 K=3 + relu -> rows 24..31 ====
  for (int i = 0; i < 4; ++i) {
    int l = lane + 32 * i;
    float x0 = sOrF[l];
    float xm = (l > 0)   ? sOrF[l - 1] : 0.f;
    float xp = (l < 127) ? sOrF[l + 1] : 0.f;
#pragma unroll
    for (int o = 0; o < 8; ++o) {
      float d = op_w[o * 3 + 0] * xm + op_w[o * 3 + 1] * x0 +
                op_w[o * 3 + 2] * xp + op_b[o];
      sFeat[(24 + o) * LN_ + l] = fmaxf(d, 0.f);
    }
  }
  __syncthreads();

  // ================= attention fusion + pooling + head =====================
  float pooled[8];
#pragma unroll
  for (int o = 0; o < 8; ++o) pooled[o] = 0.f;
  for (int i = 0; i < 4; ++i) {
    int l = lane + 32 * i;
    float sc[4];
#pragma unroll
    for (int j = 0; j < 4; ++j) {
      float s = ap_b[j];
      for (int c = 0; c < CS_; ++c) s += ap_w[j * CS_ + c] * sFeat[c * LN_ + l];
      sc[j] = s;
    }
    float mx = fmaxf(fmaxf(sc[0], sc[1]), fmaxf(sc[2], sc[3]));
    float e0 = __expf(sc[0] - mx), e1 = __expf(sc[1] - mx);
    float e2 = __expf(sc[2] - mx), e3 = __expf(sc[3] - mx);
    float inv = 1.f / (e0 + e1 + e2 + e3);
    float w0 = e0 * inv, w1 = e1 * inv, w2 = e2 * inv, w3 = e3 * inv;
#pragma unroll
    for (int c = 0; c < 8; ++c) {
      float ws = sFeat[c * LN_ + l] * w0 + sFeat[(8 + c) * LN_ + l] * w1 +
                 sFeat[(16 + c) * LN_ + l] * w2 + sFeat[(24 + c) * LN_ + l] * w3;
      pooled[c] += ws;
    }
  }
#pragma unroll
  for (int c = 0; c < 8; ++c) {
#pragma unroll
    for (int off = 16; off > 0; off >>= 1) pooled[c] += __shfl_xor(pooled[c], off, 32);
    pooled[c] *= (1.f / 128.f);
  }
  if (lane == 0) {
    float h[4];
#pragma unroll
    for (int j = 0; j < 4; ++j) {
      float s = ol1_b[j];
#pragma unroll
      for (int c = 0; c < 8; ++c) s += ol1_w[j * 8 + c] * pooled[c];
      h[j] = fmaxf(s, 0.f);
    }
    float z = ol2_b[0];
#pragma unroll
    for (int j = 0; j < 4; ++j) z += ol2_w[j] * h[j];
    out[b] = 1.f / (1.f + __expf(-z));
  }
}

extern "C" void kernel_launch(void* const* d_in, const int* in_sizes, int n_in,
                              void* d_out, int out_size, void* d_ws, size_t ws_size,
                              hipStream_t stream) {
  (void)d_ws; (void)ws_size; (void)n_in; (void)out_size;
  const float* p[37];
  for (int i = 0; i < 37; ++i) p[i] = (const float*)d_in[i];
  const int B = in_sizes[0] / (10 * LN_);           // 8192
  fused_bdn_kernel<<<dim3(B), dim3(32), 0, stream>>>(
      p[0],  p[1],  p[2],  p[3],  p[4],
      p[5],  p[6],  p[7],  p[8],
      p[9],  p[10], p[11], p[12], p[13], p[14], p[15], p[16],
      p[17], p[18], p[19], p[20],
      p[21], p[22], p[23], p[24],
      p[25], p[26], p[27], p[28],
      p[29], p[30], p[31], p[32],
      p[33], p[34], p[35], p[36],
      (float*)d_out);
}
